// MultiheadSelfAttention_65962107732872
// MI455X (gfx1250) — matrix operand order repaired and checked by reading, not compiled, not run on hardware
//
#include <hip/hip_runtime.h>
#include <hip/hip_bf16.h>

typedef __bf16 bf16_t;
typedef __attribute__((ext_vector_type(16))) __bf16 v16bf;
typedef __attribute__((ext_vector_type(8)))  __bf16 bf16x8;
typedef __attribute__((ext_vector_type(8)))  float  v8f;

union Frag16 {            // one WMMA 16-bit A/B operand: 16 bf16 per lane
    v16bf  v;
    bf16x8 h[2];
};

// ---------------------------------------------------------------------------
// fp32 -> bf16 conversion (n must be a multiple of 4)
// ---------------------------------------------------------------------------
__global__ void f32_to_bf16_kernel(const float* __restrict__ in,
                                   bf16_t* __restrict__ out, int n) {
    int i = (blockIdx.x * blockDim.x + threadIdx.x) * 4;
    if (i >= n) return;
    float4 v = *(const float4*)(in + i);
    typedef __attribute__((ext_vector_type(4))) __bf16 bf16x4;
    bf16x4 o;
    o[0] = (bf16_t)v.x; o[1] = (bf16_t)v.y; o[2] = (bf16_t)v.z; o[3] = (bf16_t)v.w;
    *(bf16x4*)(out + i) = o;
}

// ---------------------------------------------------------------------------
// WMMA GEMM: C[M,N] = A[M,K](bf16) * W[N,K](bf16)^T, fp32 accum.
// Block = 256 threads (8 waves), wave grid 2(M) x 4(N), wave tile 32x32.
// Low register footprint (no spills); latency hidden by 8-wave occupancy.
// MODE 0: store bf16 into head layout [B,H,S,64]       (Q, K)
// MODE 1: store bf16 into transposed layout [B,H,64,S] (V)
// MODE 2: store fp32 row-major [M,N]                   (final projection)
// ---------------------------------------------------------------------------
template <int MODE>
__global__ __launch_bounds__(256)
void gemm_wmma_kernel(const bf16_t* __restrict__ A,
                      const bf16_t* __restrict__ W,
                      void* __restrict__ out,
                      int M, int N, int K, int S, int H) {
    const int lane  = threadIdx.x & 31;
    const int wave  = threadIdx.x >> 5;
    const int l16   = lane & 15;
    const int lhalf = lane >> 4;

    const int tileM0 = blockIdx.x * 64  + (wave & 1) * 32;
    const int tileN0 = blockIdx.y * 128 + (wave >> 1) * 32;

    v8f acc[2][2];
#pragma unroll
    for (int i = 0; i < 2; i++)
#pragma unroll
        for (int j = 0; j < 2; j++)
#pragma unroll
            for (int e = 0; e < 8; e++) acc[i][j][e] = 0.0f;

    for (int k0 = 0; k0 < K; k0 += 32) {
        Frag16 a[2], b[2];
#pragma unroll
        for (int i = 0; i < 2; i++) {
            const bf16_t* ap = A + (size_t)(tileM0 + i * 16 + l16) * K + k0 + lhalf * 16;
            a[i].h[0] = *(const bf16x8*)ap;
            a[i].h[1] = *(const bf16x8*)(ap + 8);
        }
#pragma unroll
        for (int j = 0; j < 2; j++) {
            const bf16_t* bp = W + (size_t)(tileN0 + j * 16 + l16) * K + k0 + lhalf * 16;
            b[j].v = *(const v16bf*)bp;
        }
#pragma unroll
        for (int i = 0; i < 2; i++)
#pragma unroll
            for (int j = 0; j < 2; j++)
                acc[i][j] = __builtin_amdgcn_wmma_f32_16x16x32_bf16(
                    false, a[i].v, false, b[j].v, (short)0, acc[i][j], false, false);
    }

#pragma unroll
    for (int i = 0; i < 2; i++)
#pragma unroll
        for (int j = 0; j < 2; j++) {
            const int n0 = tileN0 + j * 16 + l16;
#pragma unroll
            for (int r = 0; r < 8; r++) {
                const int m = tileM0 + i * 16 + r + 8 * lhalf;
                const float v = acc[i][j][r];
                if (MODE == 2) {
                    ((float*)out)[(size_t)m * N + n0] = v;
                } else {
                    const int b = m / S, s = m - b * S;
                    const int h = n0 >> 6, d = n0 & 63;
                    if (MODE == 0)
                        ((bf16_t*)out)[((((size_t)b * H + h) * S + s) << 6) + d] = (bf16_t)v;
                    else
                        ((bf16_t*)out)[(((size_t)b * H + h) * 64 + d) * (size_t)S + s] = (bf16_t)v;
                }
            }
        }
}

// ---------------------------------------------------------------------------
// RoPE in-place on [BH, S, 64] bf16, theta = 10000
// ---------------------------------------------------------------------------
__global__ void rope_kernel(bf16_t* __restrict__ X, const int* __restrict__ pos,
                            int BH, int S) {
    size_t idx = (size_t)blockIdx.x * blockDim.x + threadIdx.x;
    size_t total = (size_t)BH * S * 32;
    if (idx >= total) return;
    int    j  = (int)(idx & 31);
    size_t bs = idx >> 5;
    int    s  = (int)(bs % S);
    float  p  = (float)pos[s];
    float  freq = __powf(10000.0f, -(float)(2 * j) * (1.0f / 64.0f));
    float  sn, cs;
    __sincosf(p * freq, &sn, &cs);
    bf16_t* ptr = X + bs * 64 + 2 * j;
    float x0 = (float)ptr[0];
    float x1 = (float)ptr[1];
    ptr[0] = (bf16_t)(cs * x0 - sn * x1);
    ptr[1] = (bf16_t)(sn * x0 + cs * x1);
}

// ---------------------------------------------------------------------------
// Flash attention (causal, online softmax), one wave per 16-query tile.
// Q,K: [B,H,S,64] bf16 (RoPE applied). Vt: [B,H,64,S] bf16.
// O:   [B,S,H*64] bf16.
// Block = 256 threads = 8 waves = 128 consecutive query rows of one (b,h).
// ---------------------------------------------------------------------------
__global__ __launch_bounds__(256)
void flash_attn_kernel(const bf16_t* __restrict__ Q,
                       const bf16_t* __restrict__ Km,
                       const bf16_t* __restrict__ Vt,
                       bf16_t* __restrict__ O,
                       int H, int S) {
    __shared__ __align__(16) bf16_t ldsP[8][16][40];   // padded: 80B pitch, 16B aligned chunks

    const int lane  = threadIdx.x & 31;
    const int wave  = threadIdx.x >> 5;
    const int l16   = lane & 15;
    const int lhalf = lane >> 4;

    const int qblocks = S >> 7;
    const int bh = blockIdx.x / qblocks;
    const int q0 = (blockIdx.x % qblocks) * 128 + wave * 16;
    const int b  = bh / H;
    const int h  = bh - b * H;

    const bf16_t* Qp = Q  + (size_t)bh * S * 64;
    const bf16_t* Kp = Km + (size_t)bh * S * 64;
    const bf16_t* Vp = Vt + (size_t)bh * 64 * S;

    // Q A-fragments (rows q0..q0+15, dk split 0-31 / 32-63), reused all iterations
    Frag16 qa[2];
#pragma unroll
    for (int t = 0; t < 2; t++) {
        const bf16_t* qp = Qp + (size_t)(q0 + l16) * 64 + t * 32 + lhalf * 16;
        qa[t].h[0] = *(const bf16x8*)qp;
        qa[t].h[1] = *(const bf16x8*)(qp + 8);
    }

    v8f o[4];
#pragma unroll
    for (int t = 0; t < 4; t++)
#pragma unroll
        for (int e = 0; e < 8; e++) o[t][e] = 0.0f;

    float mrow[8], lrow[8];
#pragma unroll
    for (int r = 0; r < 8; r++) { mrow[r] = -3.0e38f; lrow[r] = 0.0f; }

    const int kEnd = q0 + 16;                 // causal bound for this tile
    for (int kk = 0; kk < kEnd; kk += 32) {
        // ---- S = Q * K^T for 32 keys (two 16-col WMMA tiles) ----
        v8f sc[2];
#pragma unroll
        for (int j = 0; j < 2; j++) {
            Frag16 kb[2];
#pragma unroll
            for (int t = 0; t < 2; t++)
                kb[t].v = *(const v16bf*)(Kp + (size_t)(kk + j * 16 + l16) * 64 + t * 32 + lhalf * 16);
            v8f z;
#pragma unroll
            for (int e = 0; e < 8; e++) z[e] = 0.0f;
#pragma unroll
            for (int t = 0; t < 2; t++)
                z = __builtin_amdgcn_wmma_f32_16x16x32_bf16(
                    false, qa[t].v, false, kb[t].v, (short)0, z, false, false);
            sc[j] = z;
        }

        // ---- scale + causal mask + per-row tile max ----
        float mt[8];
#pragma unroll
        for (int r = 0; r < 8; r++) {
            const int qrow = q0 + r + 8 * lhalf;
            float best = -3.0e38f;
#pragma unroll
            for (int j = 0; j < 2; j++) {
                const int key = kk + j * 16 + l16;
                float v = sc[j][r] * 0.125f;               // 1/sqrt(64)
                v = (key <= qrow) ? v : -3.0e38f;
                sc[j][r] = v;
                best = fmaxf(best, v);
            }
            mt[r] = best;
        }
#pragma unroll
        for (int off = 1; off < 16; off <<= 1)
#pragma unroll
            for (int r = 0; r < 8; r++)
                mt[r] = fmaxf(mt[r], __shfl_xor(mt[r], off, 32));

        // ---- online softmax update ----
        float psum[8];
#pragma unroll
        for (int r = 0; r < 8; r++) {
            const float mnew  = fmaxf(mrow[r], mt[r]);
            const float alpha = __expf(mrow[r] - mnew);
            mrow[r] = mnew;
            float ps = 0.0f;
#pragma unroll
            for (int j = 0; j < 2; j++) {
                const float p = __expf(sc[j][r] - mnew);
                sc[j][r] = p;
                ps += p;
            }
            psum[r] = ps;
            lrow[r] *= alpha;
#pragma unroll
            for (int t = 0; t < 4; t++) o[t][r] *= alpha;
        }
#pragma unroll
        for (int off = 1; off < 16; off <<= 1)
#pragma unroll
            for (int r = 0; r < 8; r++)
                psum[r] += __shfl_xor(psum[r], off, 32);
#pragma unroll
        for (int r = 0; r < 8; r++) lrow[r] += psum[r];

        // ---- C-layout -> A-layout for P via per-wave LDS staging ----
#pragma unroll
        for (int r = 0; r < 8; r++)
#pragma unroll
            for (int j = 0; j < 2; j++)
                ldsP[wave][r + 8 * lhalf][j * 16 + l16] = (bf16_t)sc[j][r];

        asm volatile("s_wait_dscnt 0" ::: "memory");

        Frag16 pa;
        pa.h[0] = *(const bf16x8*)&ldsP[wave][l16][lhalf * 16];
        pa.h[1] = *(const bf16x8*)&ldsP[wave][l16][lhalf * 16 + 8];

        // ---- O += P * V  (Vt rows are dk, contiguous along keys) ----
#pragma unroll
        for (int t = 0; t < 4; t++) {
            Frag16 vb;
            vb.v = *(const v16bf*)(Vp + (size_t)(t * 16 + l16) * S + kk + lhalf * 16);
            o[t] = __builtin_amdgcn_wmma_f32_16x16x32_bf16(
                false, pa.v, false, vb.v, (short)0, o[t], false, false);
        }
    }

    // ---- normalize and store to [B,S,H*64] bf16 ----
#pragma unroll
    for (int r = 0; r < 8; r++) {
        const float inv = 1.0f / lrow[r];
        const int sOut  = q0 + r + 8 * lhalf;
        bf16_t* op = O + ((size_t)b * S + sOut) * (size_t)(H * 64) + h * 64 + l16;
#pragma unroll
        for (int t = 0; t < 4; t++)
            op[t * 16] = (bf16_t)(o[t][r] * inv);
    }
}

// ---------------------------------------------------------------------------
// Host-side launch
// ---------------------------------------------------------------------------
extern "C" void kernel_launch(void* const* d_in, const int* in_sizes, int n_in,
                              void* d_out, int out_size, void* d_ws, size_t ws_size,
                              hipStream_t stream) {
    (void)in_sizes; (void)n_in; (void)out_size; (void)ws_size;

    const float* x   = (const float*)d_in[0];
    const int*   pos = (const int*)  d_in[1];
    const float* WQ  = (const float*)d_in[2];
    const float* WK  = (const float*)d_in[3];
    const float* WV  = (const float*)d_in[4];
    const float* WO  = (const float*)d_in[5];
    float*       out = (float*)d_out;

    const int B = 2, S = 2048, D = 1024, H = 16;
    const int M = B * S, N = D, K = D;

    bf16_t* xb = (bf16_t*)d_ws;
    bf16_t* wq = xb + (size_t)M * K;
    bf16_t* wk = wq + (size_t)N * K;
    bf16_t* wv = wk + (size_t)N * K;
    bf16_t* wo = wv + (size_t)N * K;
    bf16_t* Qh = wo + (size_t)N * K;      // [B,H,S,64]
    bf16_t* Kh = Qh + (size_t)M * D;      // [B,H,S,64]
    bf16_t* Vt = Kh + (size_t)M * D;      // [B,H,64,S]
    bf16_t* Ao = Vt + (size_t)M * D;      // [B,S,D]

    // fp32 -> bf16
    f32_to_bf16_kernel<<<(M * K / 4 + 255) / 256, 256, 0, stream>>>(x, xb, M * K);
    f32_to_bf16_kernel<<<(N * K / 4 + 255) / 256, 256, 0, stream>>>(WQ, wq, N * K);
    f32_to_bf16_kernel<<<(N * K / 4 + 255) / 256, 256, 0, stream>>>(WK, wk, N * K);
    f32_to_bf16_kernel<<<(N * K / 4 + 255) / 256, 256, 0, stream>>>(WV, wv, N * K);
    f32_to_bf16_kernel<<<(N * K / 4 + 255) / 256, 256, 0, stream>>>(WO, wo, N * K);

    dim3 gg(M / 64, N / 128), gb(256);
    gemm_wmma_kernel<0><<<gg, gb, 0, stream>>>(xb, wq, Qh, M, N, K, S, H);
    gemm_wmma_kernel<0><<<gg, gb, 0, stream>>>(xb, wk, Kh, M, N, K, S, H);
    gemm_wmma_kernel<1><<<gg, gb, 0, stream>>>(xb, wv, Vt, M, N, K, S, H);

    const size_t ropeTotal = (size_t)B * H * S * 32;
    rope_kernel<<<(int)((ropeTotal + 255) / 256), 256, 0, stream>>>(Qh, pos, B * H, S);
    rope_kernel<<<(int)((ropeTotal + 255) / 256), 256, 0, stream>>>(Kh, pos, B * H, S);

    flash_attn_kernel<<<B * H * (S / 128), 256, 0, stream>>>(Qh, Kh, Vt, Ao, H, S);

    gemm_wmma_kernel<2><<<gg, gb, 0, stream>>>(Ao, wo, out, M, N, K, S, H);
}
